// FixedNet_45157286150863
// MI455X (gfx1250) — matrix operand order repaired and checked by reading, not compiled, not run on hardware
//
#include <hip/hip_runtime.h>

typedef __attribute__((ext_vector_type(16))) __bf16 v16bf;
typedef __attribute__((ext_vector_type(8)))  float  v8f;

union FragBF { v16bf v; uint4 u[2]; };

__device__ __forceinline__ __bf16 f2bf(float f) {
  unsigned u = __builtin_bit_cast(unsigned, f);
  u += 0x7FFFu + ((u >> 16) & 1u);
  unsigned short s = (unsigned short)(u >> 16);
  return __builtin_bit_cast(__bf16, s);
}
__device__ __forceinline__ float bf2f(__bf16 x) {
  unsigned short s = __builtin_bit_cast(unsigned short, x);
  return __builtin_bit_cast(float, ((unsigned)s) << 16);
}

// ---------------------------------------------------------------------------
// bf16 WMMA GEMM:  Out[b][M][N] = A[b][M][K] * B + bias, *scale
//   MODE 0: B[k*ldb + n]   (plain row-major K x N)
//   MODE 1: implicit-GEMM conv3x3, reflect pad; B = X[Ci][H][W], K = Ci*9
//   MODE 2: B[n*ldb + k]   (transposed; covariance A * A^T)
// Block: 256 threads = 8 waves in a 2(M) x 4(N) grid -> 32x64 C macro-tile.
// K stepped by 64 through LDS; 2 WMMAs (v_wmma_f32_16x16x32_bf16) per wave
// per step. Fast vectorized staging when the tile is full, guarded otherwise.
// ---------------------------------------------------------------------------
template <int MODE>
__global__ void __launch_bounds__(256)
gemm_wmma(const __bf16* __restrict__ A, long aStrideB,
          const __bf16* __restrict__ Bm, long bStrideB,
          const float* __restrict__ bias, int biasStrideB,
          void* __restrict__ Out, long oStrideB,
          int M, int N, int K, int lda, int ldb,
          int H, int W, float scale, int relu, int outBf16)
{
  __shared__ __align__(16) __bf16 sA[32 * 72];   // [m][k], pad 72 (144B rows)
  __shared__ __align__(16) __bf16 sB[64 * 72];   // [n][k] (transposed store)

  const int bb = blockIdx.z;
  const __bf16* Ab = A + (long)bb * aStrideB;
  const __bf16* Bb = Bm + (long)bb * bStrideB;
  const int n0 = blockIdx.x * 64;
  const int m0 = blockIdx.y * 32;
  const int tid  = threadIdx.x;
  const int lane = tid & 31;
  const int wv   = tid >> 5;
  const int mi   = wv >> 2;        // 0..1
  const int ni   = wv & 3;         // 0..3
  const __bf16 ZB = __builtin_bit_cast(__bf16, (unsigned short)0);

  const bool nfull = (n0 + 64 <= N);
  const bool mfull = (m0 + 32 <= M);

  v8f acc = {0.f, 0.f, 0.f, 0.f, 0.f, 0.f, 0.f, 0.f};

  for (int k0 = 0; k0 < K; k0 += 64) {
    const bool kfull = (k0 + 64 <= K);

    // ---------------- stage A tile: 32 x 64 ----------------
    {
      const int r = tid >> 3, cg = (tid & 7) << 3;
      if (kfull && mfull) {
        if (k0 + 128 <= K)
          __builtin_prefetch(&Ab[(long)(m0 + r) * lda + k0 + 64 + cg], 0, 1);
        *(uint4*)&sA[r * 72 + cg] = *(const uint4*)&Ab[(long)(m0 + r) * lda + k0 + cg];
      } else {
#pragma unroll
        for (int j = 0; j < 8; ++j) {
          int m = m0 + r, k = k0 + cg + j;
          sA[r * 72 + cg + j] = (m < M && k < K) ? Ab[(long)m * lda + k] : ZB;
        }
      }
    }

    // ---------------- stage B tile: 64(K) x 64(N) -> sB[n][k] ----------------
    if (MODE == 1) {                     // conv3x3 gather, reflect pad
      if (kfull) {
#pragma unroll
        for (int i = 0; i < 16; ++i) {
          int flat = i * 256 + tid;
          int kk = flat >> 6, nn = flat & 63;
          int k = k0 + kk, n = n0 + nn;
          int ci = k / 9, t = k - ci * 9;
          int ky = t / 3, kx = t - ky * 3;
          int h = n / W, w = n - h * W;
          int hh = h + ky - 1;
          hh = hh < 0 ? -hh : (hh >= H ? 2 * H - 2 - hh : hh);
          int ww = w + kx - 1;
          ww = ww < 0 ? -ww : (ww >= W ? 2 * W - 2 - ww : ww);
          sB[nn * 72 + kk] = Bb[((long)ci * H + hh) * W + ww];
        }
      } else {
#pragma unroll
        for (int i = 0; i < 16; ++i) {
          int flat = i * 256 + tid;
          int kk = flat >> 6, nn = flat & 63;
          int k = k0 + kk, n = n0 + nn;
          __bf16 v = ZB;
          if (k < K && n < N) {
            int ci = k / 9, t = k - ci * 9;
            int ky = t / 3, kx = t - ky * 3;
            int h = n / W, w = n - h * W;
            int hh = h + ky - 1;
            hh = hh < 0 ? -hh : (hh >= H ? 2 * H - 2 - hh : hh);
            int ww = w + kx - 1;
            ww = ww < 0 ? -ww : (ww >= W ? 2 * W - 2 - ww : ww);
            v = Bb[((long)ci * H + hh) * W + ww];
          }
          sB[nn * 72 + kk] = v;
        }
      }
    } else if (MODE == 0) {              // plain K x N
      if (kfull && nfull) {
#pragma unroll
        for (int i = 0; i < 2; ++i) {
          int kk = (tid >> 3) + i * 32, cg = (tid & 7) << 3;
          if (k0 + 128 <= K)
            __builtin_prefetch(&Bb[(long)(k0 + 64 + kk) * ldb + n0 + cg], 0, 1);
          uint4 d = *(const uint4*)&Bb[(long)(k0 + kk) * ldb + n0 + cg];
          const __bf16* dv = (const __bf16*)&d;
#pragma unroll
          for (int j = 0; j < 8; ++j) sB[(cg + j) * 72 + kk] = dv[j];
        }
      } else {
#pragma unroll
        for (int i = 0; i < 16; ++i) {
          int flat = i * 256 + tid;
          int kk = flat >> 6, nn = flat & 63;
          int k = k0 + kk, n = n0 + nn;
          sB[nn * 72 + kk] = (k < K && n < N) ? Bb[(long)k * ldb + n] : ZB;
        }
      }
    } else {                              // MODE 2: B[n*ldb + k] (contig along k)
      if (kfull && nfull) {
#pragma unroll
        for (int i = 0; i < 2; ++i) {
          int chunk = i * 256 + tid;
          int nn = chunk >> 3, kg = (chunk & 7) << 3;
          *(uint4*)&sB[nn * 72 + kg] =
              *(const uint4*)&Bb[(long)(n0 + nn) * ldb + k0 + kg];
        }
      } else {
#pragma unroll
        for (int i = 0; i < 16; ++i) {
          int flat = i * 256 + tid;
          int kk = flat >> 6, nn = flat & 63;
          int k = k0 + kk, n = n0 + nn;
          sB[nn * 72 + kk] = (k < K && n < N) ? Bb[(long)n * ldb + k] : ZB;
        }
      }
    }
    __syncthreads();

    // ---------------- fragments (ISA 7.12.2 layouts), 2 K-halves ----------------
    const int row = lane & 15;
    const int kh  = (lane >> 4) << 3;    // A: interleaved K pairs (0 / 8)
    const int kb  = (lane >> 4) << 3;    // B: interleaved K pairs (0 / 8), same as A
    const int ar  = (mi * 16 + row) * 72;
    const int br  = (ni * 16 + row) * 72;
    FragBF fa0, fb0, fa1, fb1;
    fa0.u[0] = *(const uint4*)&sA[ar + kh];
    fa0.u[1] = *(const uint4*)&sA[ar + 16 + kh];
    fb0.u[0] = *(const uint4*)&sB[br + kb];
    fb0.u[1] = *(const uint4*)&sB[br + 16 + kb];
    fa1.u[0] = *(const uint4*)&sA[ar + 32 + kh];
    fa1.u[1] = *(const uint4*)&sA[ar + 48 + kh];
    fb1.u[0] = *(const uint4*)&sB[br + 32 + kb];
    fb1.u[1] = *(const uint4*)&sB[br + 48 + kb];

    acc = __builtin_amdgcn_wmma_f32_16x16x32_bf16(false, fa0.v, false, fb0.v,
                                                  (short)0, acc, false, false);
    acc = __builtin_amdgcn_wmma_f32_16x16x32_bf16(false, fa1.v, false, fb1.v,
                                                  (short)0, acc, false, false);
    __syncthreads();
  }

  // ---------------- epilogue: bias + relu + store ----------------
  const int nOut  = n0 + ni * 16 + (lane & 15);
  const int mBase = m0 + mi * 16 + ((lane >> 4) << 3);
#pragma unroll
  for (int i = 0; i < 8; ++i) {
    int m = mBase + i;
    if (m < M && nOut < N) {
      float v = acc[i] * scale;
      if (bias) v += bias[(long)bb * biasStrideB + m];
      if (relu) v = fmaxf(v, 0.0f);
      long o = (long)bb * oStrideB + (long)m * N + nOut;
      if (outBf16) ((__bf16*)Out)[o] = f2bf(v);
      else         ((float*)Out)[o]  = v;
    }
  }
}

// ---------------------------------------------------------------------------
// Small helper kernels
// ---------------------------------------------------------------------------
__global__ void cvt_bf16(const float* __restrict__ s, __bf16* __restrict__ d, long n) {
  long i = (long)blockIdx.x * 256 + threadIdx.x;
  if (i < n) d[i] = f2bf(s[i]);
}

__global__ void zero_f32(float* p, long n) {
  long i = (long)blockIdx.x * 256 + threadIdx.x;
  if (i < n) p[i] = 0.0f;
}

__global__ void row_mean(const float* __restrict__ src, float* __restrict__ out, int L) {
  __shared__ float sm[256];
  int r = blockIdx.x;
  const float* p = src + (long)r * L;
  float s = 0.f;
  for (int i = threadIdx.x; i < L; i += 256) s += p[i];
  sm[threadIdx.x] = s;
  __syncthreads();
  for (int st = 128; st > 0; st >>= 1) {
    if (threadIdx.x < st) sm[threadIdx.x] += sm[threadIdx.x + st];
    __syncthreads();
  }
  if (threadIdx.x == 0) out[r] = sm[0] / (float)L;
}

__global__ void sub_cvt(const float* __restrict__ x, const float* __restrict__ mean,
                        __bf16* __restrict__ out, int L, long total) {
  long i = (long)blockIdx.x * 256 + threadIdx.x;
  if (i >= total) return;
  long r = i / L;
  out[i] = f2bf(x[i] - mean[r]);
}

__global__ void pool2(const __bf16* __restrict__ in, __bf16* __restrict__ out,
                      int Hi, int Wi, long total) {
  long i = (long)blockIdx.x * 256 + threadIdx.x;
  if (i >= total) return;
  int Ho = Hi >> 1, Wo = Wi >> 1;
  int wo = (int)(i % Wo); long t = i / Wo;
  int ho = (int)(t % Ho); long ch = t / Ho;
  const __bf16* p = in + ((long)ch * Hi + ho * 2) * Wi + wo * 2;
  float m = fmaxf(fmaxf(bf2f(p[0]), bf2f(p[1])),
                  fmaxf(bf2f(p[Wi]), bf2f(p[Wi + 1])));
  out[i] = f2bf(m);
}

__global__ void up2x(const __bf16* __restrict__ in, __bf16* __restrict__ out,
                     int Hi, int Wi, long total) {
  long i = (long)blockIdx.x * 256 + threadIdx.x;
  if (i >= total) return;
  int Ho = Hi * 2, Wo = Wi * 2;
  int w = (int)(i % Wo); long t = i / Wo;
  int h = (int)(t % Ho); long ch = t / Ho;
  out[i] = in[((long)ch * Hi + (h >> 1)) * Wi + (w >> 1)];
}

// out[b][o] = sum_k v[b][k] * Wt[o][k] + bias[o]     (1024x1024 matvec)
__global__ void matvec1024(const float* __restrict__ v, const float* __restrict__ Wt,
                           const float* __restrict__ b, float* __restrict__ out) {
  int o  = blockIdx.x * 256 + threadIdx.x;
  int bb = blockIdx.y;
  const float* vb = v + (long)bb * 1024;
  const float* wr = Wt + (long)o * 1024;
  float s = 0.f;
  for (int k = 0; k < 1024; ++k) s += vb[k] * wr[k];
  out[(long)bb * 1024 + o] = s + b[o];
}

// trans[b] = sMat[b] @ cMat[b]  (32x32), output bf16
__global__ void mm32(const float* __restrict__ S, const float* __restrict__ Cm,
                     __bf16* __restrict__ T) {
  int bb = blockIdx.x;
  int r = threadIdx.x >> 5, c = threadIdx.x & 31;
  const float* Sb = S + (long)bb * 1024;
  const float* Cb = Cm + (long)bb * 1024;
  float s = 0.f;
  for (int k = 0; k < 32; ++k) s += Sb[r * 32 + k] * Cb[k * 32 + c];
  T[(long)bb * 1024 + r * 32 + c] = f2bf(s);
}

__global__ void make_fbias(const float* __restrict__ uzb, const float* __restrict__ sMean,
                           float* __restrict__ fb) {
  int i = blockIdx.x * 256 + threadIdx.x;   // i in [0, 512)
  fb[i] = uzb[i & 255] + sMean[i];
}

__global__ void scat_den(const float* __restrict__ pts, float* __restrict__ den,
                         float* __restrict__ wout, int* __restrict__ hwout,
                         int Np, int H, int W) {
  long i = (long)blockIdx.x * 256 + threadIdx.x;
  if (i >= (long)2 * Np) return;
  int b = (int)(i / Np);
  const float* p = pts + i * 3;
  float px = p[0], py = p[1], pz = p[2];
  int ix = (int)((px + 1.0f) * 0.5f * W); ix = ix < 0 ? 0 : (ix > W - 1 ? W - 1 : ix);
  int iy = (int)((py + 1.0f) * 0.5f * H); iy = iy < 0 ? 0 : (iy > H - 1 ? H - 1 : iy);
  float w = __expf(-10.0f * pz);
  int hw = iy * W + ix;
  wout[i] = w;
  hwout[i] = hw;
  atomicAdd(&den[(long)b * H * W + hw], w);
}

__global__ void scat_num(const __bf16* __restrict__ fT, const float* __restrict__ wgt,
                         const int* __restrict__ hwi, float* __restrict__ num,
                         int Np, int C, int HW) {
  long i = (long)blockIdx.x * 256 + threadIdx.x;
  long total = (long)2 * C * Np;
  if (i >= total) return;
  int n = (int)(i % Np); long t = i / Np;
  int c = (int)(t % C);  int b = (int)(t / C);
  long pi = (long)b * Np + n;
  float v = bf2f(fT[i]) * wgt[pi];
  atomicAdd(&num[((long)b * C + c) * HW + hwi[pi]], v);
}

__global__ void norm_proj(const float* __restrict__ num, const float* __restrict__ den,
                          __bf16* __restrict__ proj, int C, int HW) {
  long i = (long)blockIdx.x * 256 + threadIdx.x;
  long total = (long)2 * C * HW;
  if (i >= total) return;
  int hw = (int)(i % HW);
  long t = i / HW;
  int b = (int)(t / C);
  proj[i] = f2bf(num[i] / (den[(long)b * HW + hw] + 1e-8f));
}

// ---------------------------------------------------------------------------
extern "C" void kernel_launch(void* const* d_in, const int* in_sizes, int n_in,
                              void* d_out, int out_size, void* d_ws, size_t ws_size,
                              hipStream_t stream) {
  const float* style  = (const float*)d_in[0];
  const float* points = (const float*)d_in[1];
  const float* feats  = (const float*)d_in[2];
  const float *ew0=(const float*)d_in[3],  *eb0=(const float*)d_in[4];
  const float *ew1=(const float*)d_in[5],  *eb1=(const float*)d_in[6];
  const float *ew2=(const float*)d_in[7],  *eb2=(const float*)d_in[8];
  const float *ew3=(const float*)d_in[9],  *eb3=(const float*)d_in[10];
  const float *ew4=(const float*)d_in[11], *eb4=(const float*)d_in[12];
  const float *ew5=(const float*)d_in[13], *eb5=(const float*)d_in[14];
  const float *dw0=(const float*)d_in[15], *db0=(const float*)d_in[16];
  const float *dw1=(const float*)d_in[17], *db1=(const float*)d_in[18];
  const float *dw2=(const float*)d_in[19], *db2=(const float*)d_in[20];
  const float *dw3=(const float*)d_in[21], *db3=(const float*)d_in[22];
  const float *dw4=(const float*)d_in[23], *db4=(const float*)d_in[24];
  const float *comp_w=(const float*)d_in[25], *comp_b=(const float*)d_in[26];
  const float *unzip_w=(const float*)d_in[27], *unzip_b=(const float*)d_in[28];
  const float *cw0=(const float*)d_in[29], *cb0=(const float*)d_in[30];
  const float *cw1=(const float*)d_in[31], *cb1=(const float*)d_in[32];
  const float *cw2=(const float*)d_in[33], *cb2=(const float*)d_in[34];
  const float *cfc_w=(const float*)d_in[35], *cfc_b=(const float*)d_in[36];
  const float *sw0=(const float*)d_in[37], *sb0=(const float*)d_in[38];
  const float *sw1=(const float*)d_in[39], *sb1=(const float*)d_in[40];
  const float *sw2=(const float*)d_in[41], *sb2=(const float*)d_in[42];
  const float *sfc_w=(const float*)d_in[43], *sfc_b=(const float*)d_in[44];

  const long NP = 65536;          // points per batch
  const int  Hc = 128, Wc = 128;  // splat canvas

  // ---- bump allocator over d_ws ----
  size_t off = 0;
  auto alloc = [&](size_t bytes) -> void* {
    off = (off + 255) & ~(size_t)255;
    void* p = (char*)d_ws + off;
    off += bytes;
    return p;
  };
  auto g1 = [](long n) { return dim3((unsigned)((n + 255) / 256)); };
  auto cv = [&](const float* src, long n) -> __bf16* {
    __bf16* d = (__bf16*)alloc((size_t)n * 2);
    cvt_bf16<<<g1(n), 256, 0, stream>>>(src, d, n);
    return d;
  };
  auto GEMM = [&](int mode, const __bf16* A, long asb, const __bf16* Bp, long bsb,
                  const float* bias, int bsbias, void* out, long osb,
                  int M, int N, int K, int lda, int ldb,
                  int H, int W, float scale, int relu, int obf) {
    dim3 grid((N + 63) / 64, (M + 31) / 32, 2);
    if (mode == 0)
      gemm_wmma<0><<<grid, 256, 0, stream>>>(A, asb, Bp, bsb, bias, bsbias, out, osb,
                                             M, N, K, lda, ldb, H, W, scale, relu, obf);
    else if (mode == 1)
      gemm_wmma<1><<<grid, 256, 0, stream>>>(A, asb, Bp, bsb, bias, bsbias, out, osb,
                                             M, N, K, lda, ldb, H, W, scale, relu, obf);
    else
      gemm_wmma<2><<<grid, 256, 0, stream>>>(A, asb, Bp, bsb, bias, bsbias, out, osb,
                                             M, N, K, lda, ldb, H, W, scale, relu, obf);
  };

  // ---- convert weights to bf16 ----
  __bf16 *Wew0 = cv(ew0, 9),      *Wew1 = cv(ew1, 1728),   *Wew2 = cv(ew2, 36864);
  __bf16 *Wew3 = cv(ew3, 73728),  *Wew4 = cv(ew4, 147456), *Wew5 = cv(ew5, 294912);
  __bf16 *Wdw0 = cv(dw0, 294912), *Wdw1 = cv(dw1, 147456), *Wdw2 = cv(dw2, 73728);
  __bf16 *Wdw3 = cv(dw3, 36864),  *Wdw4 = cv(dw4, 1728);
  __bf16 *Wcomp = cv(comp_w, 8192), *Wuz = cv(unzip_w, 8192);
  __bf16 *Wcw0 = cv(cw0, 32768),  *Wcw1 = cv(cw1, 8192),   *Wcw2 = cv(cw2, 2048);
  __bf16 *Wsw0 = cv(sw0, 294912), *Wsw1 = cv(sw1, 73728),  *Wsw2 = cv(sw2, 18432);
  __bf16 *styb = cv(style, 2L * 3 * 65536);

  // ---- encoder: style -> sF [2,256,64,64] ----
  __bf16* e0 = (__bf16*)alloc(2L * 3 * 65536 * 2);
  GEMM(0, Wew0, 0, styb, 3L * 65536, eb0, 0, e0, 3L * 65536, 3, 65536, 3, 3, 65536, 0, 0, 1.f, 0, 1);
  __bf16* e1 = (__bf16*)alloc(2L * 64 * 65536 * 2);
  GEMM(1, Wew1, 0, e0, 3L * 65536, eb1, 0, e1, 64L * 65536, 64, 65536, 27, 27, 0, 256, 256, 1.f, 1, 1);
  __bf16* e2 = (__bf16*)alloc(2L * 64 * 65536 * 2);
  GEMM(1, Wew2, 0, e1, 64L * 65536, eb2, 0, e2, 64L * 65536, 64, 65536, 576, 576, 0, 256, 256, 1.f, 1, 1);
  __bf16* p1 = (__bf16*)alloc(2L * 64 * 16384 * 2);
  pool2<<<g1(2L * 64 * 16384), 256, 0, stream>>>(e2, p1, 256, 256, 2L * 64 * 16384);
  __bf16* e3 = (__bf16*)alloc(2L * 128 * 16384 * 2);
  GEMM(1, Wew3, 0, p1, 64L * 16384, eb3, 0, e3, 128L * 16384, 128, 16384, 576, 576, 0, 128, 128, 1.f, 1, 1);
  __bf16* e4 = (__bf16*)alloc(2L * 128 * 16384 * 2);
  GEMM(1, Wew4, 0, e3, 128L * 16384, eb4, 0, e4, 128L * 16384, 128, 16384, 1152, 1152, 0, 128, 128, 1.f, 1, 1);
  __bf16* p2 = (__bf16*)alloc(2L * 128 * 4096 * 2);
  pool2<<<g1(2L * 128 * 4096), 256, 0, stream>>>(e4, p2, 128, 128, 2L * 128 * 4096);
  float* sF = (float*)alloc(2L * 256 * 4096 * 4);
  GEMM(1, Wew5, 0, p2, 128L * 4096, eb5, 0, sF, 256L * 4096, 256, 4096, 1152, 1152, 0, 64, 64, 1.f, 1, 0);

  float* sMean = (float*)alloc(512 * 4);
  row_mean<<<512, 256, 0, stream>>>(sF, sMean, 4096);
  __bf16* sFc = (__bf16*)alloc(2L * 256 * 4096 * 2);
  sub_cvt<<<g1(2L * 256 * 4096), 256, 0, stream>>>(sF, sMean, sFc, 4096, 2L * 256 * 4096);

  // ---- style covariance branch -> sMat ----
  __bf16* gg1 = (__bf16*)alloc(2L * 128 * 4096 * 2);
  GEMM(1, Wsw0, 0, sFc, 256L * 4096, sb0, 0, gg1, 128L * 4096, 128, 4096, 2304, 2304, 0, 64, 64, 1.f, 1, 1);
  __bf16* gg2 = (__bf16*)alloc(2L * 64 * 4096 * 2);
  GEMM(1, Wsw1, 0, gg1, 128L * 4096, sb1, 0, gg2, 64L * 4096, 64, 4096, 1152, 1152, 0, 64, 64, 1.f, 1, 1);
  __bf16* gg = (__bf16*)alloc(2L * 32 * 4096 * 2);
  GEMM(1, Wsw2, 0, gg2, 64L * 4096, sb2, 0, gg, 32L * 4096, 32, 4096, 576, 576, 0, 64, 64, 1.f, 0, 1);
  float* covs = (float*)alloc(2L * 1024 * 4);
  GEMM(2, gg, 32L * 4096, gg, 32L * 4096, nullptr, 0, covs, 1024, 32, 32, 4096, 4096, 4096,
       0, 0, 1.0f / 4096.0f, 0, 0);
  float* sMat = (float*)alloc(2L * 1024 * 4);
  matvec1024<<<dim3(4, 2), 256, 0, stream>>>(covs, sfc_w, sfc_b, sMat);

  // ---- point feature branch -> cMat, comp ----
  float* cMean = (float*)alloc(512 * 4);
  row_mean<<<512, 256, 0, stream>>>(feats, cMean, 65536);
  __bf16* cFc = (__bf16*)alloc(2L * 256 * NP * 2);              // 67 MB slab (reused later)
  sub_cvt<<<g1(2L * 256 * NP), 256, 0, stream>>>(feats, cMean, cFc, 65536, 2L * 256 * NP);
  __bf16* comp = (__bf16*)alloc(2L * 32 * NP * 2);
  GEMM(0, Wcomp, 0, cFc, 256L * NP, comp_b, 0, comp, 32L * NP, 32, 65536, 256, 256, 65536, 0, 0, 1.f, 0, 1);
  __bf16* h1 = (__bf16*)alloc(2L * 128 * NP * 2);               // 33.5 MB slab (reused later)
  GEMM(0, Wcw0, 0, cFc, 256L * NP, cb0, 0, h1, 128L * NP, 128, 65536, 256, 256, 65536, 0, 0, 1.f, 1, 1);
  __bf16* h2 = (__bf16*)alloc(2L * 64 * NP * 2);                // 16.8 MB slab (reused later)
  GEMM(0, Wcw1, 0, h1, 128L * NP, cb1, 0, h2, 64L * NP, 64, 65536, 128, 128, 65536, 0, 0, 1.f, 1, 1);
  __bf16* hh = (__bf16*)alloc(2L * 32 * NP * 2);
  GEMM(0, Wcw2, 0, h2, 64L * NP, cb2, 0, hh, 32L * NP, 32, 65536, 64, 64, 65536, 0, 0, 1.f, 0, 1);
  float* covc = (float*)alloc(2L * 1024 * 4);
  GEMM(2, hh, 32L * NP, hh, 32L * NP, nullptr, 0, covc, 1024, 32, 32, 65536, 65536, 65536,
       0, 0, 1.0f / 65536.0f, 0, 0);
  float* cMat = (float*)alloc(2L * 1024 * 4);
  matvec1024<<<dim3(4, 2), 256, 0, stream>>>(covc, cfc_w, cfc_b, cMat);

  // ---- trans = sMat @ cMat ; tf = trans @ comp ; featsT = unzip(tf)+bias ----
  __bf16* transb = (__bf16*)alloc(2L * 1024 * 2);
  mm32<<<2, 1024, 0, stream>>>(sMat, cMat, transb);
  __bf16* tf = (__bf16*)alloc(2L * 32 * NP * 2);
  GEMM(0, transb, 1024, comp, 32L * NP, nullptr, 0, tf, 32L * NP, 32, 65536, 32, 32, 65536, 0, 0, 1.f, 0, 1);
  float* fbias = (float*)alloc(512 * 4);
  make_fbias<<<2, 256, 0, stream>>>(unzip_b, sMean, fbias);
  __bf16* fT = cFc;                                             // reuse: cFc dead
  GEMM(0, Wuz, 0, tf, 32L * NP, fbias, 256, fT, 256L * NP, 256, 65536, 32, 32, 65536, 0, 0, 1.f, 0, 1);

  // ---- point splatter into canvas ----
  float* num = (float*)alloc(2L * 256 * 16384 * 4);             // 33.5 MB slab (reused later)
  float* den = (float*)alloc(2L * 16384 * 4);
  float* wgt = (float*)alloc(2L * NP * 4);
  int*   hwi = (int*)alloc(2L * NP * 4);
  zero_f32<<<g1(2L * 256 * 16384), 256, 0, stream>>>(num, 2L * 256 * 16384);
  zero_f32<<<g1(2L * 16384), 256, 0, stream>>>(den, 2L * 16384);
  scat_den<<<g1(2L * NP), 256, 0, stream>>>(points, den, wgt, hwi, (int)NP, Hc, Wc);
  scat_num<<<g1(2L * 256 * NP), 256, 0, stream>>>(fT, wgt, hwi, num, (int)NP, 256, Hc * Wc);
  __bf16* proj = (__bf16*)alloc(2L * 256 * 16384 * 2);
  norm_proj<<<g1(2L * 256 * 16384), 256, 0, stream>>>(num, den, proj, 256, 16384);

  // ---- decoder: proj [2,256,128,128] -> out [2,3,512,512] ----
  __bf16* y0 = (__bf16*)alloc(2L * 128 * 16384 * 2);
  GEMM(1, Wdw0, 0, proj, 256L * 16384, db0, 0, y0, 128L * 16384, 128, 16384, 2304, 2304, 0, 128, 128, 1.f, 1, 1);
  __bf16* y0u = h1;                                             // reuse 33.5 MB slab
  up2x<<<g1(2L * 128 * 65536), 256, 0, stream>>>(y0, y0u, 128, 128, 2L * 128 * 65536);
  __bf16* y1 = cFc;                                             // reuse 67 MB slab (fT dead)
  GEMM(1, Wdw1, 0, y0u, 128L * 65536, db1, 0, y1, 128L * 65536, 128, 65536, 1152, 1152, 0, 256, 256, 1.f, 1, 1);
  __bf16* y2 = h2;                                              // reuse 16.8 MB slab
  GEMM(1, Wdw2, 0, y1, 128L * 65536, db2, 0, y2, 64L * 65536, 64, 65536, 1152, 1152, 0, 256, 256, 1.f, 1, 1);
  __bf16* y2u = (__bf16*)num;                                   // reuse 33.5 MB slab (num dead)
  up2x<<<g1(2L * 64 * 262144), 256, 0, stream>>>(y2, y2u, 256, 256, 2L * 64 * 262144);
  __bf16* y3 = cFc;                                             // reuse 67 MB slab (y1 dead)
  GEMM(1, Wdw3, 0, y2u, 64L * 262144, db3, 0, y3, 64L * 262144, 64, 262144, 576, 576, 0, 512, 512, 1.f, 1, 1);
  GEMM(1, Wdw4, 0, y3, 64L * 262144, db4, 0, d_out, 3L * 262144, 3, 262144, 576, 576, 0, 512, 512, 1.f, 0, 0);
}
